// GPT_37641093382446
// MI455X (gfx1250) — matrix operand order repaired and checked by reading, not compiled, not run on hardware
//
#include <hip/hip_runtime.h>
#include <math.h>

// Model constants (fixed by the reference)
#define Tn 2048
#define Dn 512
#define Hn 8
#define Lx 2
#define TD (Tn * Dn)

// ---- WMMA fragment machinery (gfx1250 wave32, 16x16x32 bf16) -------------
typedef __attribute__((ext_vector_type(16))) __bf16 v16bf;
typedef __attribute__((ext_vector_type(8)))  float  v8f;
typedef __attribute__((ext_vector_type(4)))  unsigned int tdm_g0_t;
typedef __attribute__((ext_vector_type(8)))  int          tdm_g1_t;
typedef __attribute__((ext_vector_type(4)))  int          tdm_g2_t;
typedef __attribute__((ext_vector_type(8)))  int          tdm_g4_t;

struct alignas(16) U128 { unsigned int w[4]; };
union BFrag { v16bf v; unsigned short u[16]; U128 q[2]; };

// float -> bf16, round-to-nearest-even (bit-level; storage is ushort)
__device__ inline unsigned short f2bf(float f) {
  unsigned int u = __float_as_uint(f);
  u += 0x7FFFu + ((u >> 16) & 1u);
  return (unsigned short)(u >> 16);
}

__device__ inline v8f wmma_bf16(const BFrag& a, const BFrag& b, v8f c) {
  // D = A(16x32) * B(32x16) + C, fp32 accumulate
  return __builtin_amdgcn_wmma_f32_16x16x32_bf16(
      false, a.v, false, b.v, (short)0, c, false, false);
}

// Fragment layouts (order used here, wave32; A and B share it):
//  A 16x32 bf16: lane m(<16) row=m; elem e -> k = base16 + e,
//                base16 = (lane>=16)?16:0. => one contiguous 16-elem (32B) run.
//  B 32x16 bf16: lane n(&15) col=n; elem e -> k = base16 + e,
//                base16 = (lane>=16)?16:0. => one contiguous 16-elem (32B) run.
//  C/D 16x16 f32: elem e -> row = e + ((lane>=16)?8:0), col = lane&15.

// TDM 2D tile load: rows x 64 bf16 tile from row-major [.,K] matrix -> LDS.
// D# packing per CDNA5 ISA:
//  g0: [0]=count=1, [1]=lds_addr, [2]=gaddr[31:0], [3]=gaddr[56:32]|type=2<<30
//  g1: [0]=data_size(2B)<<16, dim0@48, dim1@80, tile0@112, tile1@128,
//      tile2@144(=0), stride0@160, stride1@208(=0, 2D)
// This toolchain exposes the 6-arg builtin (g0, g1, g2, g3, g4, cpol).
__device__ inline void tdm_load_tile(const unsigned short* gsrc,
                                     unsigned int lds_off, int K, int rows) {
  unsigned long long ga = (unsigned long long)(size_t)gsrc;
  tdm_g0_t g0;
  g0[0] = 1u;                                     // count=1, user descriptor
  g0[1] = lds_off;                                // LDS byte address
  g0[2] = (unsigned int)ga;                       // global addr lo
  g0[3] = (unsigned int)((ga >> 32) & 0x01FFFFFFull) | (2u << 30); // hi|type=2
  tdm_g1_t g1;
  g1[0] = (int)(1u << 16);                        // data_size code 1 (2 bytes)
  g1[1] = (int)(((unsigned)K & 0xFFFFu) << 16);   // tensor_dim0[15:0]
  g1[2] = (int)((((unsigned)K >> 16) & 0xFFFFu) | ((unsigned)Tn << 16));
  g1[3] = (int)(64u << 16);                       // dim1 hi=0 | tile_dim0=64
  g1[4] = (int)(unsigned)rows;                    // tile_dim1 | tile_dim2=0
  g1[5] = (int)(unsigned)K;                       // tensor_dim0_stride lo
  g1[6] = 0;                                      // stride0 hi | stride1 lo
  g1[7] = 0;
  tdm_g2_t z4 = {0, 0, 0, 0};
  tdm_g4_t z8 = {0, 0, 0, 0, 0, 0, 0, 0};
  __builtin_amdgcn_tensor_load_to_lds(g0, g1, z4, z4, z8, 0);
}

// ---- small elementwise kernels -------------------------------------------
// fp32 [G][K][N] -> bf16 transposed [G][N][K]
__global__ void cvtT_k(const float* __restrict__ src,
                       unsigned short* __restrict__ dst, int K, int N, int n) {
  int i = blockIdx.x * 256 + threadIdx.x;
  if (i >= n) return;
  int slice = K * N;
  int g = i / slice, r = i % slice;
  int nn = r / K, kk = r % K;
  dst[i] = f2bf(src[(size_t)g * slice + (size_t)kk * N + nn]);
}

__global__ void addpos_k(const float* __restrict__ x,
                         const float* __restrict__ pos,
                         float* __restrict__ h0) {
  int i = blockIdx.x * 256 + threadIdx.x;
  if (i < TD) h0[i] = x[i] + pos[i];   // B==1, pos is [T,D]
}

// LayerNorm over D=512 with 256 threads (2 elems/thread).
__global__ __launch_bounds__(256)
void layernorm_k(const float* __restrict__ x, const float* __restrict__ w,
                 const float* __restrict__ b, unsigned short* outB, float* outF) {
  __shared__ float red[256];
  const int row = blockIdx.x;
  const int t = threadIdx.x;
  const float* xr = x + (size_t)row * Dn;
  float a0 = xr[t], a1 = xr[t + 256];
  red[t] = a0 + a1;
  __syncthreads();
  for (int o = 128; o > 0; o >>= 1) { if (t < o) red[t] += red[t + o]; __syncthreads(); }
  float mu = red[0] * (1.0f / Dn);
  __syncthreads();
  float d0 = a0 - mu, d1 = a1 - mu;
  red[t] = d0 * d0 + d1 * d1;
  __syncthreads();
  for (int o = 128; o > 0; o >>= 1) { if (t < o) red[t] += red[t + o]; __syncthreads(); }
  float rstd = rsqrtf(red[0] * (1.0f / Dn) + 1e-5f);
  float y0 = d0 * rstd * w[t] + b[t];
  float y1 = d1 * rstd * w[t + 256] + b[t + 256];
  if (outB) {
    outB[(size_t)row * Dn + t]       = f2bf(y0);
    outB[(size_t)row * Dn + t + 256] = f2bf(y1);
  } else {
    outF[(size_t)row * Dn + t]       = y0;
    outF[(size_t)row * Dn + t + 256] = y1;
  }
}

// ---- bf16 WMMA GEMM: C[M,N] = A[M,K]@W^T + bias, fused epilogue ----------
// WT is pre-transposed bf16 [N][K] so B-fragments load contiguously from
// global. A tiles (128x64) stream through LDS via TDM, double buffered.
// mode 0: outB = bf16(C); mode 1: outF = C + res; mode 2: outB = bf16(gelu(C))
__global__ __launch_bounds__(256)
void gemm_bf16_k(const unsigned short* __restrict__ A,
                 const unsigned short* __restrict__ WT,
                 const float* __restrict__ bias, const float* __restrict__ res,
                 float* __restrict__ outF, unsigned short* __restrict__ outB,
                 int N, int K, int mode) {
  __shared__ __align__(16) unsigned short As[2][128][64];  // 32KB double buffer
  const int tid  = threadIdx.x;
  const int lane = tid & 31;
  const int wave = tid >> 5;
  const int wm = wave >> 1, wn = wave & 1;                 // 4x2 wave grid
  const int m0 = blockIdx.x * 128;
  const int n0 = blockIdx.y * 64;
  const int l15 = lane & 15;
  const int half8  = (lane >> 4) << 3;
  const int half16 = (lane >> 4) << 4;

  v8f acc[2][2] = {};

  // prologue: DMA first A tile (wave 0 only; TDM ignores EXEC, per-wave issue)
  if (tid == 0)
    tdm_load_tile(&A[(size_t)m0 * K], (unsigned int)(size_t)&As[0][0][0], K, 128);

  int buf = 0;
  for (int k0 = 0; k0 < K; k0 += 64, buf ^= 1) {
    __builtin_amdgcn_s_wait_tensorcnt(0);   // wave0: tile arrived; others: nop
    __syncthreads();                        // publish tile, retire prior reads
    if (tid == 0 && k0 + 64 < K)            // overlap next DMA with compute
      tdm_load_tile(&A[(size_t)m0 * K + k0 + 64],
                    (unsigned int)(size_t)&As[buf ^ 1][0][0], K, 128);
    // prefetch next B k-chunk into cache while this one computes
    if (k0 + 64 < K)
      __builtin_prefetch(&WT[(size_t)(n0 + (tid & 63)) * K + k0 + 64], 0, 3);

#pragma unroll
    for (int kk = 0; kk < 64; kk += 32) {
      BFrag a[2], b[2];
#pragma unroll
      for (int mt = 0; mt < 2; ++mt) {
        int row = wm * 32 + mt * 16 + l15;
        a[mt].q[0] = *(const U128*)&As[buf][row][kk + half16];
        a[mt].q[1] = *(const U128*)&As[buf][row][kk + half16 + 8];
      }
#pragma unroll
      for (int nt = 0; nt < 2; ++nt) {
        const unsigned short* bp =
            WT + (size_t)(n0 + wn * 32 + nt * 16 + l15) * K + k0 + kk + half16;
        b[nt].q[0] = *(const U128*)bp;
        b[nt].q[1] = *(const U128*)(bp + 8);
      }
#pragma unroll
      for (int mt = 0; mt < 2; ++mt)
#pragma unroll
        for (int nt = 0; nt < 2; ++nt)
          acc[mt][nt] = wmma_bf16(a[mt], b[nt], acc[mt][nt]);
    }
    __syncthreads();                        // done reading As[buf]
  }

#pragma unroll
  for (int mt = 0; mt < 2; ++mt) {
#pragma unroll
    for (int nt = 0; nt < 2; ++nt) {
      int col = n0 + wn * 32 + nt * 16 + l15;
      float bv = bias[col];
#pragma unroll
      for (int e = 0; e < 8; ++e) {
        int row = m0 + wm * 32 + mt * 16 + half8 + e;
        float v = acc[mt][nt][e] + bv;
        size_t o = (size_t)row * N + col;
        if (mode == 1)      outF[o] = v + res[o];
        else if (mode == 2) outB[o] = f2bf(0.5f * v * (1.0f + erff(v * 0.70710678118f)));
        else                outB[o] = f2bf(v);
      }
    }
  }
}

// ---- banded-causal flash attention, one wave per (head, 16-query tile) ---
// Q,K,V,Y: bf16 [T, D] with head h occupying columns [h*64, h*64+64).
__global__ __launch_bounds__(32)
void attn_flash_k(const unsigned short* __restrict__ Q,
                  const unsigned short* __restrict__ Km,
                  const unsigned short* __restrict__ V,
                  unsigned short* __restrict__ Y, int bw) {
  __shared__ __align__(16) unsigned short Pst[16][32];   // P re-layout C->A
  __shared__ __align__(16) unsigned short Vst[32][64];   // staged V block
  const int lane = threadIdx.x;
  const int qi0  = blockIdx.x * 16;
  const int hoff = blockIdx.y * 64;
  const int l15 = lane & 15;
  const int half8  = (lane >> 4) << 3;
  const int half16 = (lane >> 4) << 4;

  // Q A-fragments (Dh=64 -> two K=32 chunks), contiguous 16B global loads
  BFrag qf[2];
#pragma unroll
  for (int dc = 0; dc < 2; ++dc) {
    const unsigned short* src =
        Q + (size_t)(qi0 + l15) * Dn + hoff + dc * 32 + half16;
    qf[dc].q[0] = *(const U128*)src;
    qf[dc].q[1] = *(const U128*)(src + 8);
  }

  v8f O[4] = {};
  float mrow[8], lrow[8];
#pragma unroll
  for (int e = 0; e < 8; ++e) { mrow[e] = -3e38f; lrow[e] = 0.0f; }

  int jlo = qi0 - bw; if (jlo < 0) jlo = 0;
  jlo &= ~31;                       // align to 32-key blocks (mask handles extras)
  const int jhi = qi0 + 16;

  for (int j0 = jlo; j0 < jhi; j0 += 32) {
    __syncthreads();                // protect Pst/Vst reuse across iterations
    // prefetch next K/V block while this one computes
    if (j0 + 32 < jhi) {
      __builtin_prefetch(&Km[(size_t)(j0 + 32 + lane) * Dn + hoff], 0, 3);
      __builtin_prefetch(&V[(size_t)(j0 + 32 + lane) * Dn + hoff], 0, 3);
    }
    // stage V[j0..j0+31][hoff..hoff+63] -> LDS (coalesced b128)
#pragma unroll
    for (int it = 0; it < 8; ++it) {
      int c = it * 32 + lane;
      int kr = c >> 3, dc8 = (c & 7) << 3;
      *(U128*)&Vst[kr][dc8] =
          *(const U128*)&V[(size_t)(j0 + kr) * Dn + hoff + dc8];
    }

    // S = Q * K^T  (two 16x16 tiles, each accumulated over 2 K=32 chunks)
    v8f S[2] = {};
#pragma unroll
    for (int t = 0; t < 2; ++t) {
#pragma unroll
      for (int dc = 0; dc < 2; ++dc) {
        BFrag kf;                   // B-frag of K^T: per lane 16 contiguous d
        const unsigned short* ks =
            Km + (size_t)(j0 + 16 * t + l15) * Dn + hoff + dc * 32 + half16;
        kf.q[0] = *(const U128*)ks;
        kf.q[1] = *(const U128*)(ks + 8);
        S[t] = wmma_bf16(qf[dc], kf, S[t]);
      }
    }

    // band mask + scale + online softmax
    float p0[8], p1[8], mblk[8], rs[8], alpha[8];
#pragma unroll
    for (int e = 0; e < 8; ++e) {
      int i = qi0 + half8 + e;
      int ja = j0 + l15, jb = j0 + 16 + l15;
      float sa = ((ja <= i) && (ja >= i - bw)) ? S[0][e] * 0.125f : -3e38f;
      float sb = ((jb <= i) && (jb >= i - bw)) ? S[1][e] * 0.125f : -3e38f;
      p0[e] = sa; p1[e] = sb;
      mblk[e] = fmaxf(sa, sb);
    }
#pragma unroll
    for (int off = 1; off < 16; off <<= 1)
#pragma unroll
      for (int e = 0; e < 8; ++e)
        mblk[e] = fmaxf(mblk[e], __shfl_xor(mblk[e], off, 32));
#pragma unroll
    for (int e = 0; e < 8; ++e) {
      float mnew = fmaxf(mrow[e], mblk[e]);
      alpha[e] = (mrow[e] <= -1e37f) ? 0.0f : expf(mrow[e] - mnew);
      float pa = (p0[e] <= -1e37f) ? 0.0f : expf(p0[e] - mnew);
      float pb = (p1[e] <= -1e37f) ? 0.0f : expf(p1[e] - mnew);
      p0[e] = pa; p1[e] = pb;
      rs[e] = pa + pb;
      mrow[e] = mnew;
    }
#pragma unroll
    for (int off = 1; off < 16; off <<= 1)
#pragma unroll
      for (int e = 0; e < 8; ++e)
        rs[e] += __shfl_xor(rs[e], off, 32);
#pragma unroll
    for (int e = 0; e < 8; ++e) {
      lrow[e] = lrow[e] * alpha[e] + rs[e];
#pragma unroll
      for (int f = 0; f < 4; ++f) O[f][e] *= alpha[e];
      Pst[half8 + e][l15]      = f2bf(p0[e]);   // C-layout -> LDS
      Pst[half8 + e][16 + l15] = f2bf(p1[e]);
    }
    __syncthreads();

    // P as A-fragment (16x32), V as B-fragments, O += P*V
    BFrag pf;
    pf.q[0] = *(const U128*)&Pst[l15][half16];
    pf.q[1] = *(const U128*)&Pst[l15][half16 + 8];
#pragma unroll
    for (int f = 0; f < 4; ++f) {
      BFrag vf;
#pragma unroll
      for (int e = 0; e < 16; ++e)
        vf.u[e] = Vst[half16 + e][f * 16 + l15];
      O[f] = wmma_bf16(pf, vf, O[f]);
    }
  }

#pragma unroll
  for (int f = 0; f < 4; ++f)
#pragma unroll
    for (int e = 0; e < 8; ++e) {
      float o = O[f][e] / lrow[e];
      Y[(size_t)(qi0 + half8 + e) * Dn + hoff + f * 16 + l15] = f2bf(o);
    }
}

// ---- orchestration --------------------------------------------------------
extern "C" void kernel_launch(void* const* d_in, const int* in_sizes, int n_in,
                              void* d_out, int out_size, void* d_ws, size_t ws_size,
                              hipStream_t stream) {
  (void)in_sizes; (void)n_in; (void)out_size; (void)ws_size;
  const float* x    = (const float*)d_in[0];
  const float* pos  = (const float*)d_in[1];
  const float* ln1w = (const float*)d_in[2];
  const float* ln1b = (const float*)d_in[3];
  const float* ln2w = (const float*)d_in[4];
  const float* ln2b = (const float*)d_in[5];
  const float* Wq = (const float*)d_in[6];  const float* bq = (const float*)d_in[7];
  const float* Wk = (const float*)d_in[8];  const float* bk = (const float*)d_in[9];
  const float* Wv = (const float*)d_in[10]; const float* bv = (const float*)d_in[11];
  const float* Wo = (const float*)d_in[12]; const float* bo = (const float*)d_in[13];
  const float* W1 = (const float*)d_in[14]; const float* b1 = (const float*)d_in[15];
  const float* W2 = (const float*)d_in[16]; const float* b2 = (const float*)d_in[17];
  const float* lnfw = (const float*)d_in[18];
  const float* lnfb = (const float*)d_in[19];

  char* cur = (char*)d_ws;
  auto take = [&](size_t bytes) -> char* {
    char* p = cur;
    cur += (bytes + 255) & ~(size_t)255;
    return p;
  };
  const size_t WE  = (size_t)2 * Lx * Dn * Dn;        // D x D weight tensors
  const size_t W1E = (size_t)2 * Lx * Dn * 2 * Dn;    // D x 2D weight tensors
  unsigned short* wqB = (unsigned short*)take(WE * 2);   // all stored [N][K] bf16
  unsigned short* wkB = (unsigned short*)take(WE * 2);
  unsigned short* wvB = (unsigned short*)take(WE * 2);
  unsigned short* woB = (unsigned short*)take(WE * 2);
  unsigned short* w1B = (unsigned short*)take(W1E * 2);
  unsigned short* w2B = (unsigned short*)take(W1E * 2);
  float* h0 = (float*)take((size_t)TD * 4);
  float* h  = (float*)take((size_t)TD * 4);
  unsigned short* xln  = (unsigned short*)take((size_t)TD * 2);
  unsigned short* qb   = (unsigned short*)take((size_t)TD * 2);
  unsigned short* kb   = (unsigned short*)take((size_t)TD * 2);
  unsigned short* vb   = (unsigned short*)take((size_t)TD * 2);
  unsigned short* yb   = (unsigned short*)take((size_t)TD * 2);
  unsigned short* midb = (unsigned short*)take((size_t)Tn * 2 * Dn * 2);

  // fp32 -> bf16 weight conversion WITH transpose to [N][K] (every call)
  const unsigned gWE  = (unsigned)((WE  + 255) / 256);
  const unsigned gW1E = (unsigned)((W1E + 255) / 256);
  cvtT_k<<<dim3(gWE),  256, 0, stream>>>(Wq, wqB, Dn, Dn, (int)WE);
  cvtT_k<<<dim3(gWE),  256, 0, stream>>>(Wk, wkB, Dn, Dn, (int)WE);
  cvtT_k<<<dim3(gWE),  256, 0, stream>>>(Wv, wvB, Dn, Dn, (int)WE);
  cvtT_k<<<dim3(gWE),  256, 0, stream>>>(Wo, woB, Dn, Dn, (int)WE);
  cvtT_k<<<dim3(gW1E), 256, 0, stream>>>(W1, w1B, Dn, 2 * Dn, (int)W1E);
  cvtT_k<<<dim3(gW1E), 256, 0, stream>>>(W2, w2B, 2 * Dn, Dn, (int)W1E);

  addpos_k<<<dim3(TD / 256), 256, 0, stream>>>(x, pos, h0);

  const dim3 gD(Tn / 128, Dn / 64);
  for (int s = 0; s < 2; ++s) {
    (void)hipMemcpyAsync(h, h0, (size_t)TD * 4, hipMemcpyDeviceToDevice, stream);
    const int bw = (s == 0) ? 999 : 9;
    for (int l = 0; l < Lx; ++l) {
      const int idx = s * Lx + l;
      layernorm_k<<<Tn, 256, 0, stream>>>(h, ln1w + idx * Dn, ln1b + idx * Dn, xln, nullptr);
      gemm_bf16_k<<<gD, 256, 0, stream>>>(xln, wqB + (size_t)idx * Dn * Dn, bq + idx * Dn,
                                          nullptr, nullptr, qb, Dn, Dn, 0);
      gemm_bf16_k<<<gD, 256, 0, stream>>>(xln, wkB + (size_t)idx * Dn * Dn, bk + idx * Dn,
                                          nullptr, nullptr, kb, Dn, Dn, 0);
      gemm_bf16_k<<<gD, 256, 0, stream>>>(xln, wvB + (size_t)idx * Dn * Dn, bv + idx * Dn,
                                          nullptr, nullptr, vb, Dn, Dn, 0);
      attn_flash_k<<<dim3(Tn / 16, Hn), 32, 0, stream>>>(qb, kb, vb, yb, bw);
      gemm_bf16_k<<<gD, 256, 0, stream>>>(yb, woB + (size_t)idx * Dn * Dn, bo + idx * Dn,
                                          h, h, nullptr, Dn, Dn, 1);
      layernorm_k<<<Tn, 256, 0, stream>>>(h, ln2w + idx * Dn, ln2b + idx * Dn, xln, nullptr);
      gemm_bf16_k<<<dim3(Tn / 128, (2 * Dn) / 64), 256, 0, stream>>>(
          xln, w1B + (size_t)idx * Dn * 2 * Dn, b1 + idx * 2 * Dn,
          nullptr, nullptr, midb, 2 * Dn, Dn, 2);
      gemm_bf16_k<<<gD, 256, 0, stream>>>(midb, w2B + (size_t)idx * 2 * Dn * Dn, b2 + idx * Dn,
                                          h, h, nullptr, Dn, 2 * Dn, 1);
    }
    layernorm_k<<<Tn, 256, 0, stream>>>(h, lnfw, lnfb, nullptr,
                                        (float*)d_out + (size_t)s * TD);
  }
}
